// CrossAttention_40802189312391
// MI455X (gfx1250) — matrix operand order repaired and checked by reading, not compiled, not run on hardware
//
#include <hip/hip_runtime.h>
#include <hip/hip_bf16.h>

// ---------------------------------------------------------------------------
// Cross-attention for MI455X (gfx1250): bf16 WMMA pipeline.
// ---------------------------------------------------------------------------

typedef __bf16 bf16;
typedef bf16  v8bf  __attribute__((ext_vector_type(8)));
typedef bf16  v16bf __attribute__((ext_vector_type(16)));
typedef float v8f   __attribute__((ext_vector_type(8)));

#define BATCH  2
#define NSEQ   2048
#define N1SEQ  2048
#define DIMC   1024
#define QDIM   768
#define TDIM   3072      // 3*DIM
#define HEADS  16
#define HD     64
#define ATTN_SCALE 0.125f  // HEAD_DIM^-0.5

// ---- fragment loaders -----------------------------------------------------
// A-matrix 16x32 (16-bit): lane<16 -> row=lane, K = {0..7} and {16..23}
//                          lane>=16 -> row=lane-16, K = {8..15} and {24..31}
__device__ __forceinline__ v16bf frag_a(const bf16* base, int ld, int r0, int k0, int lane) {
    const bf16* p = base + (size_t)(r0 + (lane & 15)) * ld + (size_t)(k0 + ((lane >> 4) << 3));
    v8bf lo = *(const v8bf*)p;
    v8bf hi = *(const v8bf*)(p + 16);
    return __builtin_shufflevector(lo, hi, 0,1,2,3,4,5,6,7,8,9,10,11,12,13,14,15);
}

// B-matrix 32x16 (16-bit): lane holds column N=lane%16, K = 8*(lane/16)+{0..7} and 16+8*(lane/16)+{0..7}
// (same as A). Our "B" operands are stored row-major [N, K] (weight rows /
// key-token rows / V^T rows), so one output column == one contiguous row.
__device__ __forceinline__ v16bf frag_b(const bf16* base, int ld, int c0, int k0, int lane) {
    const bf16* p = base + (size_t)(c0 + (lane & 15)) * ld + (size_t)(k0 + ((lane >> 4) << 3));
    v8bf lo = *(const v8bf*)p;
    v8bf hi = *(const v8bf*)(p + 16);
    return __builtin_shufflevector(lo, hi, 0,1,2,3,4,5,6,7,8,9,10,11,12,13,14,15);
}

__device__ __forceinline__ v8f wmma_bf16(v16bf a, v16bf b, v8f c) {
    return __builtin_amdgcn_wmma_f32_16x16x32_bf16(false, a, false, b, (short)0, c, false, false);
}

// ---- elementwise f32 -> bf16 ----------------------------------------------
__global__ void xattn_f32_to_bf16(const float* __restrict__ s, bf16* __restrict__ d, int n) {
    int i = blockIdx.x * blockDim.x + threadIdx.x;
    if (i < n) d[i] = (bf16)s[i];
}

// ---- V^T builder: vt[b,h,d,m] = qkv_y[b,m, 2*DIM + h*HD + d] ---------------
__global__ void xattn_build_vt(const bf16* __restrict__ qkvy, bf16* __restrict__ vt) {
    size_t i = (size_t)blockIdx.x * blockDim.x + threadIdx.x;  // over B*H*HD*N1
    int m = (int)(i % N1SEQ);
    size_t r = i / N1SEQ;
    int d = (int)(r % HD);  r /= HD;
    int h = (int)(r % HEADS);
    int b = (int)(r / HEADS);
    vt[i] = qkvy[((size_t)(b * N1SEQ + m)) * TDIM + 2 * DIMC + h * HD + d];
}

// ---- generic NT GEMM: C[M,N] = A[M,K] * W[N,K]^T (+bias) -------------------
// grid = (N/128, M/64), block = 128 (4 waves); wave tile 32x64 = 2x4 WMMA tiles
__global__ __launch_bounds__(128) void xattn_gemm_nt(
    const bf16* __restrict__ A, int lda,
    const bf16* __restrict__ W, int ldb,
    float* __restrict__ Cf, bf16* __restrict__ Cb, int ldc,
    int K, const float* __restrict__ bias) {
    const int lane = threadIdx.x & 31;
    const int wave = threadIdx.x >> 5;
    const int row0 = blockIdx.y * 64 + (wave >> 1) * 32;
    const int col0 = blockIdx.x * 128 + (wave & 1) * 64;

    v8f z = {};
    v8f acc[2][4];
#pragma unroll
    for (int mi = 0; mi < 2; ++mi)
#pragma unroll
        for (int ni = 0; ni < 4; ++ni) acc[mi][ni] = z;

    for (int k0 = 0; k0 < K; k0 += 32) {
        v16bf af[2], bfv[4];
#pragma unroll
        for (int mi = 0; mi < 2; ++mi) af[mi] = frag_a(A, lda, row0 + 16 * mi, k0, lane);
#pragma unroll
        for (int ni = 0; ni < 4; ++ni) bfv[ni] = frag_b(W, ldb, col0 + 16 * ni, k0, lane);
        if (k0 + 32 < K) {  // L2 prefetch of next A tile (global_prefetch_b8)
            __builtin_prefetch(A + (size_t)(row0 + (lane & 15)) * lda + k0 + 32, 0, 1);
        }
#pragma unroll
        for (int mi = 0; mi < 2; ++mi)
#pragma unroll
            for (int ni = 0; ni < 4; ++ni)
                acc[mi][ni] = wmma_bf16(af[mi], bfv[ni], acc[mi][ni]);
    }

    // C/D layout: vgpr j, lane L -> row = j + 8*(L/16), col = L%16
    const int hi = lane >> 4, nn = lane & 15;
#pragma unroll
    for (int mi = 0; mi < 2; ++mi)
#pragma unroll
        for (int ni = 0; ni < 4; ++ni)
#pragma unroll
            for (int j = 0; j < 8; ++j) {
                int r = row0 + 16 * mi + j + 8 * hi;
                int c = col0 + 16 * ni + nn;
                float v = acc[mi][ni][j];
                if (bias) v += bias[c];
                if (Cf) Cf[(size_t)r * ldc + c] = v;
                else    Cb[(size_t)r * ldc + c] = (bf16)v;
            }
}

// ---- flash attention: one wave per 16 query rows per (b,h) -----------------
// grid = (N/64, HEADS, BATCH), block = 128 (4 waves, wave w -> rows +16*w)
__global__ __launch_bounds__(128) void xattn_flash(
    const bf16* __restrict__ qkvx,   // [B*N, 3*DIM]  (q at offset 0)
    const bf16* __restrict__ qkvy,   // [B*N1, 3*DIM] (k at offset DIM)
    const bf16* __restrict__ vt,     // [B, H, HD, N1] (V transposed)
    bf16* __restrict__ attn_out) {   // [B*N, DIM]
    const int lane = threadIdx.x & 31;
    const int wave = threadIdx.x >> 5;
    const int h = blockIdx.y;
    const int b = blockIdx.z;
    const int row0 = blockIdx.x * 64 + wave * 16;

    __shared__ __align__(16) bf16 Pl[4][16 * 32];   // per-wave P staging tile
    bf16* pp = &Pl[wave][0];

    const bf16* qbase = qkvx + (size_t)b * NSEQ * TDIM + h * HD;
    const bf16* kbase = qkvy + (size_t)b * N1SEQ * TDIM + DIMC + h * HD;
    const bf16* vbase = vt + ((size_t)(b * HEADS + h)) * HD * N1SEQ;

    // resident Q fragments (K-dim = HD = 64 -> 2 WMMA k-steps)
    v16bf qf0 = frag_a(qbase, TDIM, row0, 0, lane);
    v16bf qf1 = frag_a(qbase, TDIM, row0, 32, lane);

    float mrow[8], lrow[8];
#pragma unroll
    for (int j = 0; j < 8; ++j) { mrow[j] = -1e30f; lrow[j] = 0.0f; }
    v8f z = {};
    v8f o[4];
#pragma unroll
    for (int t = 0; t < 4; ++t) o[t] = z;

    const int hi = lane >> 4, nn = lane & 15;

    for (int c0 = 0; c0 < N1SEQ; c0 += 32) {
        // --- S = (Q K^T) * scale for 32 key columns (two 16x16 C tiles) ---
        v8f s0 = z, s1 = z;
        v16bf k00 = frag_b(kbase, TDIM, c0,      0,  lane);
        v16bf k01 = frag_b(kbase, TDIM, c0,      32, lane);
        v16bf k10 = frag_b(kbase, TDIM, c0 + 16, 0,  lane);
        v16bf k11 = frag_b(kbase, TDIM, c0 + 16, 32, lane);
        s0 = wmma_bf16(qf0, k00, s0); s0 = wmma_bf16(qf1, k01, s0);
        s1 = wmma_bf16(qf0, k10, s1); s1 = wmma_bf16(qf1, k11, s1);

        // --- online softmax in C layout (row j+8*hi lives in one 16-lane half)
        float alf[8];
#pragma unroll
        for (int j = 0; j < 8; ++j) {
            float a = s0[j] * ATTN_SCALE;
            float c = s1[j] * ATTN_SCALE;
            float mx = fmaxf(a, c);
            mx = fmaxf(mx, __shfl_xor(mx, 1, 32));
            mx = fmaxf(mx, __shfl_xor(mx, 2, 32));
            mx = fmaxf(mx, __shfl_xor(mx, 4, 32));
            mx = fmaxf(mx, __shfl_xor(mx, 8, 32));
            float mnew = fmaxf(mrow[j], mx);
            float al = __expf(mrow[j] - mnew);
            float p0 = __expf(a - mnew);
            float p1 = __expf(c - mnew);
            float rs = p0 + p1;
            rs += __shfl_xor(rs, 1, 32);
            rs += __shfl_xor(rs, 2, 32);
            rs += __shfl_xor(rs, 4, 32);
            rs += __shfl_xor(rs, 8, 32);
            lrow[j] = lrow[j] * al + rs;
            mrow[j] = mnew;
            alf[j] = al;
            int r = j + 8 * hi;                 // C-layout -> LDS row-major P[16][32]
            pp[r * 32 + nn]      = (bf16)p0;
            pp[r * 32 + 16 + nn] = (bf16)p1;
        }
#pragma unroll
        for (int t = 0; t < 4; ++t)
#pragma unroll
            for (int j = 0; j < 8; ++j) o[t][j] *= alf[j];

        // --- reload P in A-fragment layout (DS ops are in-order per wave) ---
        v16bf pf = frag_a(pp, 32, 0, 0, lane);

        // --- O += P * V  (V^T rows are the B-fragment pattern, K = 32) -----
#pragma unroll
        for (int t = 0; t < 4; ++t) {
            v16bf vf = frag_b(vbase, N1SEQ, t * 16, c0, lane);
            o[t] = wmma_bf16(pf, vf, o[t]);
        }
    }

    // --- normalize and store [B,N,DIM] with head offset ---------------------
#pragma unroll
    for (int t = 0; t < 4; ++t)
#pragma unroll
        for (int j = 0; j < 8; ++j) {
            int r = row0 + j + 8 * hi;
            float v = o[t][j] / lrow[j];
            attn_out[((size_t)(b * NSEQ + r)) * DIMC + h * HD + t * 16 + nn] = (bf16)v;
        }
}

// ---------------------------------------------------------------------------
extern "C" void kernel_launch(void* const* d_in, const int* in_sizes, int n_in,
                              void* d_out, int out_size, void* d_ws, size_t ws_size,
                              hipStream_t stream) {
    (void)in_sizes; (void)n_in; (void)out_size; (void)ws_size;
    const float* x      = (const float*)d_in[0];
    const float* y      = (const float*)d_in[1];
    const float* w_q    = (const float*)d_in[2];
    const float* w_qkv  = (const float*)d_in[3];
    const float* w_proj = (const float*)d_in[4];
    const float* b_proj = (const float*)d_in[5];
    float* out = (float*)d_out;

    // workspace carve (bf16 buffers, 256B aligned)
    char* ws = (char*)d_ws;
    auto carve = [&](size_t elems) -> bf16* {
        bf16* p = (bf16*)ws;
        ws += ((elems * sizeof(bf16)) + 255) & ~(size_t)255;
        return p;
    };
    const int nx     = BATCH * NSEQ * DIMC;    // 4194304
    const int ny     = BATCH * N1SEQ * QDIM;   // 3145728
    const int nwq    = DIMC * QDIM;            // 786432
    const int nwqkv  = TDIM * DIMC;            // 3145728
    const int nwproj = DIMC * DIMC;            // 1048576
    bf16* xb       = carve(nx);
    bf16* yb       = carve(ny);
    bf16* wqb      = carve(nwq);
    bf16* wqkvb    = carve(nwqkv);
    bf16* wprojb   = carve(nwproj);
    bf16* yqb      = carve((size_t)BATCH * N1SEQ * DIMC);
    bf16* qkvx     = carve((size_t)BATCH * NSEQ * TDIM);
    bf16* qkvy     = carve((size_t)BATCH * N1SEQ * TDIM);
    bf16* vtb      = carve((size_t)BATCH * HEADS * HD * N1SEQ);
    bf16* attn_out = carve((size_t)BATCH * NSEQ * DIMC);

    // 1) convert inputs/weights to bf16
    xattn_f32_to_bf16<<<(nx + 255) / 256, 256, 0, stream>>>(x, xb, nx);
    xattn_f32_to_bf16<<<(ny + 255) / 256, 256, 0, stream>>>(y, yb, ny);
    xattn_f32_to_bf16<<<(nwq + 255) / 256, 256, 0, stream>>>(w_q, wqb, nwq);
    xattn_f32_to_bf16<<<(nwqkv + 255) / 256, 256, 0, stream>>>(w_qkv, wqkvb, nwqkv);
    xattn_f32_to_bf16<<<(nwproj + 255) / 256, 256, 0, stream>>>(w_proj, wprojb, nwproj);

    // 2) yq = y @ w_q^T        [4096,768] x [1024,768]^T -> bf16 [4096,1024]
    dim3 g1(DIMC / 128, (BATCH * N1SEQ) / 64);
    xattn_gemm_nt<<<g1, 128, 0, stream>>>(yb, QDIM, wqb, QDIM,
                                          nullptr, yqb, DIMC, QDIM, nullptr);

    // 3) qkv_x = x @ w_qkv^T   [4096,1024] x [3072,1024]^T -> bf16 [4096,3072]
    dim3 g2(TDIM / 128, (BATCH * NSEQ) / 64);
    xattn_gemm_nt<<<g2, 128, 0, stream>>>(xb, DIMC, wqkvb, DIMC,
                                          nullptr, qkvx, TDIM, DIMC, nullptr);

    // 4) qkv_y = yq @ w_qkv^T
    xattn_gemm_nt<<<g2, 128, 0, stream>>>(yqb, DIMC, wqkvb, DIMC,
                                          nullptr, qkvy, TDIM, DIMC, nullptr);

    // 5) V^T for the PV matmul
    xattn_build_vt<<<(BATCH * HEADS * HD * N1SEQ) / 256, 256, 0, stream>>>(qkvy, vtb);

    // 6) flash attention -> attn_out bf16 [4096,1024]
    dim3 ga(NSEQ / 64, HEADS, BATCH);
    xattn_flash<<<ga, 128, 0, stream>>>(qkvx, qkvy, vtb, attn_out);

    // 7) out = attn_out @ w_proj^T + b_proj  -> fp32 d_out
    dim3 g4(DIMC / 128, (BATCH * NSEQ) / 64);
    xattn_gemm_nt<<<g4, 128, 0, stream>>>(attn_out, DIMC, wprojb, DIMC,
                                          out, nullptr, DIMC, DIMC, b_proj);
}
